// RepLKMatching_40879498729309
// MI455X (gfx1250) — matrix operand order repaired and checked by reading, not compiled, not run on hardware
//
#include <hip/hip_runtime.h>
#include <hip/hip_bf16.h>

// ---------------- problem constants (from reference) ----------------
#define BB 2
#define CC 128
#define HH 48
#define WW 160
#define DD 96
#define HW (HH*WW)
#define KIN 224          // C + D input channels to the conv
#define KTOT (KIN*9)     // 2016 GEMM K
#define NKC (KTOT/32)    // 63 K-chunks of 32
#define MIN_DEPTH 0.1f
#define DEPTH_STEP (19.9f/95.0f)

typedef __attribute__((ext_vector_type(8)))  float          v8f;
typedef __attribute__((ext_vector_type(8)))  unsigned short v8u;
typedef __attribute__((ext_vector_type(16))) unsigned short v16u;
typedef __attribute__((ext_vector_type(16))) __bf16         v16bf;

// ---- bf16 helpers (round-to-nearest-even) and hi/lo error split ----
__device__ __forceinline__ unsigned short f2bf(float x) {
  unsigned int u = __float_as_uint(x);
  unsigned int r = u + 0x7FFFu + ((u >> 16) & 1u);
  return (unsigned short)(r >> 16);
}
__device__ __forceinline__ float bf2f(unsigned short h) {
  return __uint_as_float(((unsigned int)h) << 16);
}
__device__ __forceinline__ void bfsplit(float x, unsigned short& hi, unsigned short& lo) {
  hi = f2bf(x);
  lo = f2bf(x - bf2f(hi));
}

// ---- gfx1250 async global->LDS copy (ASYNCcnt path), with fallback ----
// Probe-learned signature: (int4 addrspace(1)*, int4 addrspace(3)*, imm, cpol)
typedef __attribute__((__vector_size__(16))) int i4_t;
typedef __attribute__((address_space(1))) i4_t gv4i;
typedef __attribute__((address_space(3))) i4_t lv4i;

#if defined(__has_builtin)
#if __has_builtin(__builtin_amdgcn_global_load_async_to_lds_b128)
#define HAVE_ASYNC_LDS 1
#endif
#endif

__device__ __forceinline__ void async_copy16(const void* g, void* l) {
#ifdef HAVE_ASYNC_LDS
  // GLOBAL_LOAD_ASYNC_TO_LDS_B128: LDS[l] = MEM[g], tracked by ASYNCcnt.
  __builtin_amdgcn_global_load_async_to_lds_b128((gv4i*)g, (lv4i*)l, 0, 0);
#else
  *(v8u*)l = *(const v8u*)g;
#endif
}

__device__ __forceinline__ void wait_async_lds() {
#ifdef HAVE_ASYNC_LDS
#if __has_builtin(__builtin_amdgcn_s_wait_asynccnt)
  __builtin_amdgcn_s_wait_asynccnt(0);
#else
  asm volatile("s_wait_asynccnt 0x0" ::: "memory");
#endif
#endif
}

// ====================================================================
// Kernel 1: plane-sweep cost volume.  One thread per (b,d,h,w).
// Gathers are L2-resident (lookup_feats = 7.9 MB << 192 MB L2).
// ====================================================================
__global__ void k1_plane_sweep(const float* __restrict__ cur,
                               const float* __restrict__ lut,
                               const float* __restrict__ poses,
                               const float* __restrict__ Km,
                               const float* __restrict__ invK,
                               float* __restrict__ costRaw) {
  int idx = blockIdx.x * blockDim.x + threadIdx.x;
  if (idx >= BB * DD * HH * WW) return;
  int w = idx % WW; int t = idx / WW;
  int h = t % HH;   t /= HH;
  int d = t % DD;   int b = t / DD;

  const float* kb = Km   + b * 16;
  const float* ik = invK + b * 16;
  const float* ps = poses + b * 16;   // F==1

  float fx = (float)w, fy = (float)h;
  // cam = invK[:3,:3] @ (x,y,1)
  float cx = ik[0] * fx + ik[1] * fy + ik[2];
  float cy = ik[4] * fx + ik[5] * fy + ik[6];
  float cz = ik[8] * fx + ik[9] * fy + ik[10];
  float depth = MIN_DEPTH + (float)d * DEPTH_STEP;
  float wx = depth * cx, wy = depth * cy, wz = depth * cz;

  // P = (K @ pose)[:3]
  float P[12];
#pragma unroll
  for (int i = 0; i < 3; ++i)
#pragma unroll
    for (int j = 0; j < 4; ++j) {
      float s = 0.f;
#pragma unroll
      for (int k = 0; k < 4; ++k) s += kb[i * 4 + k] * ps[k * 4 + j];
      P[i * 4 + j] = s;
    }
  float uu = P[0] * wx + P[1] * wy + P[2]  * wz + P[3];
  float vv = P[4] * wx + P[5] * wy + P[6]  * wz + P[7];
  float zz = P[8] * wx + P[9] * wy + P[10] * wz + P[11];
  float gx = uu / (zz + 1e-7f);
  float gy = vv / (zz + 1e-7f);

  float x0f = floorf(gx), y0f = floorf(gy);
  float ax = gx - x0f, ay = gy - y0f;
  int x0 = (int)x0f, y0 = (int)y0f, x1 = x0 + 1, y1 = y0 + 1;
  float w00 = (1.f - ax) * (1.f - ay), w10 = ax * (1.f - ay);
  float w01 = (1.f - ax) * ay,         w11 = ax * ay;
  w00 *= (x0 >= 0 && x0 < WW && y0 >= 0 && y0 < HH) ? 1.f : 0.f;
  w10 *= (x1 >= 0 && x1 < WW && y0 >= 0 && y0 < HH) ? 1.f : 0.f;
  w01 *= (x0 >= 0 && x0 < WW && y1 >= 0 && y1 < HH) ? 1.f : 0.f;
  w11 *= (x1 >= 0 && x1 < WW && y1 >= 0 && y1 < HH) ? 1.f : 0.f;
  int xc0 = min(max(x0, 0), WW - 1), xc1 = min(max(x1, 0), WW - 1);
  int yc0 = min(max(y0, 0), HH - 1), yc1 = min(max(y1, 0), HH - 1);
  int i00 = yc0 * WW + xc0, i10 = yc0 * WW + xc1;
  int i01 = yc1 * WW + xc0, i11 = yc1 * WW + xc1;

  const float* lfb = lut + (size_t)b * CC * HW;
  const float* cfb = cur + (size_t)b * CC * HW + h * WW + w;
  float acc = 0.f;
#pragma unroll 4
  for (int c = 0; c < CC; ++c) {
    const float* lf = lfb + c * HW;
    float s = w00 * lf[i00] + w10 * lf[i10] + w01 * lf[i01] + w11 * lf[i11];
    acc += fabsf(s - cfb[c * HW]);
  }
  float meand = acc * (1.0f / CC);

  float edge = (gx >= 2.f && gx <= (float)(WW - 2) &&
                gy >= 2.f && gy <= (float)(HH - 2) &&
                h >= 2 && h < HH - 2 && w >= 2 && w < WW - 2) ? 1.f : 0.f;
  float psum = 0.f;
#pragma unroll
  for (int i = 0; i < 16; ++i) psum += ps[i];
  float vp = (psum != 0.f) ? 1.f : 0.f;

  float diff = meand * edge * vp;
  float cnt  = (diff > 0.f) ? 1.f : 0.f;
  costRaw[idx] = diff / (cnt + 1e-7f);
}

// ====================================================================
// Kernel 2: per-pixel depth reduction (missing-fill, conf, argmin) and
// channel-contiguous bf16 hi/lo repack of concat(feats, cost*conf).
// ====================================================================
__global__ void k2_reduce_pack(const float* __restrict__ costRaw,
                               const float* __restrict__ cur,
                               float* __restrict__ outLowest,
                               float* __restrict__ outConf,
                               unsigned short* __restrict__ catHi,
                               unsigned short* __restrict__ catLo) {
  int idx = blockIdx.x * blockDim.x + threadIdx.x;
  if (idx >= BB * HW) return;
  int b = idx / HW; int hw = idx - b * HW;
  const float* cr = costRaw + (size_t)b * DD * HW + hw;

  float maxc = 0.f;
  bool allpos = true;
  for (int d = 0; d < DD; ++d) {
    float c = cr[d * HW];
    maxc = fmaxf(maxc, c);
    allpos = allpos && (c > 0.f);
  }
  float conf = allpos ? 1.f : 0.f;

  unsigned short* ch = catHi + (size_t)idx * KIN;
  unsigned short* cl = catLo + (size_t)idx * KIN;
  float best = 3.4e38f; int bestd = 0;
  for (int d = 0; d < DD; ++d) {
    float c = cr[d * HW];
    float adj = (c == 0.f) ? maxc : c;
    float viz = (adj == 0.f) ? 100.f : adj;
    if (viz < best) { best = viz; bestd = d; }
    unsigned short hi, lo;
    bfsplit(adj * conf, hi, lo);
    ch[CC + d] = hi; cl[CC + d] = lo;
  }
  for (int c = 0; c < CC; ++c) {
    unsigned short hi, lo;
    bfsplit(cur[(size_t)(b * CC + c) * HW + hw], hi, lo);
    ch[c] = hi; cl[c] = lo;
  }
  outLowest[idx] = 1.0f / (MIN_DEPTH + (float)bestd * DEPTH_STEP);
  outConf[idx] = conf;
}

// ====================================================================
// Kernel 3: repack conv weights into per-lane WMMA B-fragment order.
// GEMM K index: k = tap*224 + ic (tap = ky*3+kx) so a 32-chunk never
// crosses a tap.  B fragment (bf16 32x16): lane%16 = N col, reg j = K (table order).
// dst[((kc*8+nt)*32+lane)*16 + j] = w[oc = nt*16+lane%16][k = kc*32+((j&8)<<1)+(lane>>4)*8+(j&7)]
// ====================================================================
__global__ void k3_pack_weights(const float* __restrict__ cw,
                                unsigned short* __restrict__ wHi,
                                unsigned short* __restrict__ wLo) {
  int t = blockIdx.x * blockDim.x + threadIdx.x;
  if (t >= NKC * 8 * 32 * 16) return;
  int j    = t & 15;
  int lane = (t >> 4) & 31;
  int nt   = (t >> 9) & 7;
  int kc   = t >> 12;
  int k    = kc * 32 + ((j & 8) << 1) + ((lane >> 4) << 3) + (j & 7);
  int tap  = k / KIN;
  int ic   = k - tap * KIN;
  int ky   = tap / 3, kx = tap - ky * 3;
  int oc   = nt * 16 + (lane & 15);
  float wv = cw[(((size_t)oc * KIN + ic) * 3 + ky) * 3 + kx];
  unsigned short hi, lo;
  bfsplit(wv, hi, lo);
  wHi[t] = hi; wLo[t] = lo;
}

// ====================================================================
// Kernel 4: implicit-GEMM 3x3 conv with v_wmma_f32_16x16x32_bf16.
// Block = one 16-pixel M-tile; 4 waves x 2 N-tiles = 128 out channels.
// Halo strip staged with GLOBAL_LOAD_ASYNC_TO_LDS_B128 (ASYNCcnt),
// out-of-bounds chunks zero-filled with ds stores (disjoint addresses).
// Error-compensated bf16: acc += Ah*Bh + Ah*Bl + Al*Bh (f32 accum).
// ====================================================================
__global__ __launch_bounds__(128)
void k4_conv_wmma(const unsigned short* __restrict__ catHi,
                  const unsigned short* __restrict__ catLo,
                  const unsigned short* __restrict__ wHi,
                  const unsigned short* __restrict__ wLo,
                  const float* __restrict__ bias,
                  float* __restrict__ out) {
  __shared__ __align__(16) unsigned short sHi[3][18][KIN];
  __shared__ __align__(16) unsigned short sLo[3][18][KIN];

  int tid = threadIdx.x;
  int p0 = blockIdx.x * 16;             // W%16==0 -> tile stays on one row
  int b = p0 / HW; int rem = p0 - b * HW;
  int y = rem / WW; int wb = rem - y * WW;

  // Stage 3-row x 18-pixel halo strip as 16-byte (8-channel) chunks.
  // A chunk's bounds depend only on (gy,gx): in-bounds chunks go through
  // the async global->LDS DMA path, halo chunks are zero ds_store_b128s.
  const int NCH = 3 * 18 * (KIN / 8);   // 1512 chunks per plane
  for (int i = tid; i < NCH; i += 128) {
    int ky = i / (18 * 28); int r = i - ky * (18 * 28);
    int px = r / 28;        int c8 = (r - px * 28) * 8;
    int gy = y + ky - 1, gx = wb + px - 1;
    unsigned short* dH = &sHi[ky][px][c8];
    unsigned short* dL = &sLo[ky][px][c8];
    if (gy >= 0 && gy < HH && gx >= 0 && gx < WW) {
      size_t g = ((size_t)(b * HH + gy) * WW + gx) * KIN + c8;
      async_copy16(catHi + g, dH);
      async_copy16(catLo + g, dL);
    } else {
      v8u z = {0, 0, 0, 0, 0, 0, 0, 0};
      *(v8u*)dH = z;
      *(v8u*)dL = z;
    }
  }
  wait_async_lds();     // drain this wave's ASYNCcnt before the barrier
  __syncthreads();      // (barrier also fences the ds zero-fill stores)

  int wave = tid >> 5, lane = tid & 31;
  int m = lane & 15, lh = lane >> 4;
  int nt0 = wave * 2, nt1 = nt0 + 1;
  v8f acc0 = {0.f, 0.f, 0.f, 0.f, 0.f, 0.f, 0.f, 0.f};
  v8f acc1 = {0.f, 0.f, 0.f, 0.f, 0.f, 0.f, 0.f, 0.f};

  for (int kc = 0; kc < NKC; ++kc) {
    int tap = kc / 7;
    int icb = (kc - tap * 7) * 32;
    int ky = tap / 3, kx = tap - ky * 3;

    // A fragment (16x32 bf16): lane m, two contiguous 8-channel runs
    // at K = lh*8.. and K = 16+lh*8..  -> two ds_load_b128 per operand.
    const unsigned short* pAh = &sHi[ky][m + kx][icb + lh * 8];
    const unsigned short* pAl = &sLo[ky][m + kx][icb + lh * 8];
    v8u ah0 = *(const v8u*)pAh;        v8u ah1 = *(const v8u*)(pAh + 16);
    v8u al0 = *(const v8u*)pAl;        v8u al1 = *(const v8u*)(pAl + 16);
    v16u ahu = __builtin_shufflevector(ah0, ah1, 0,1,2,3,4,5,6,7,8,9,10,11,12,13,14,15);
    v16u alu = __builtin_shufflevector(al0, al1, 0,1,2,3,4,5,6,7,8,9,10,11,12,13,14,15);

    // B fragments: pre-packed, 32 contiguous bytes per lane (L2-resident).
    v16u bh0u = *(const v16u*)(wHi + (((size_t)(kc * 8 + nt0) * 32 + lane) << 4));
    v16u bh1u = *(const v16u*)(wHi + (((size_t)(kc * 8 + nt1) * 32 + lane) << 4));
    v16u bl0u = *(const v16u*)(wLo + (((size_t)(kc * 8 + nt0) * 32 + lane) << 4));
    v16u bl1u = *(const v16u*)(wLo + (((size_t)(kc * 8 + nt1) * 32 + lane) << 4));

    // Prefetch next K-chunk's weight lines behind this chunk's WMMAs.
    if (kc + 1 < NKC) {
      __builtin_prefetch((const void*)(wHi + (((size_t)((kc + 1) * 8 + nt0) * 32 + lane) << 4)), 0, 0);
      __builtin_prefetch((const void*)(wLo + (((size_t)((kc + 1) * 8 + nt0) * 32 + lane) << 4)), 0, 0);
    }

    v16bf ah  = __builtin_bit_cast(v16bf, ahu);
    v16bf al  = __builtin_bit_cast(v16bf, alu);
    v16bf bh0 = __builtin_bit_cast(v16bf, bh0u);
    v16bf bh1 = __builtin_bit_cast(v16bf, bh1u);
    v16bf bl0 = __builtin_bit_cast(v16bf, bl0u);
    v16bf bl1 = __builtin_bit_cast(v16bf, bl1u);

    acc0 = __builtin_amdgcn_wmma_f32_16x16x32_bf16(false, ah, false, bh0, (short)0, acc0, false, false);
    acc0 = __builtin_amdgcn_wmma_f32_16x16x32_bf16(false, ah, false, bl0, (short)0, acc0, false, false);
    acc0 = __builtin_amdgcn_wmma_f32_16x16x32_bf16(false, al, false, bh0, (short)0, acc0, false, false);
    acc1 = __builtin_amdgcn_wmma_f32_16x16x32_bf16(false, ah, false, bh1, (short)0, acc1, false, false);
    acc1 = __builtin_amdgcn_wmma_f32_16x16x32_bf16(false, ah, false, bl1, (short)0, acc1, false, false);
    acc1 = __builtin_amdgcn_wmma_f32_16x16x32_bf16(false, al, false, bh1, (short)0, acc1, false, false);
  }

  // C/D layout: reg r -> M = r + 8*lh, lane%16 -> N.
  int oc0 = nt0 * 16 + m, oc1 = nt1 * 16 + m;
  float b0 = bias[oc0], b1 = bias[oc1];
#pragma unroll
  for (int r = 0; r < 8; ++r) {
    int x = wb + r + lh * 8;
    float v0 = acc0[r] + b0;
    float v1 = acc1[r] + b1;
    out[((size_t)(b * CC + oc0) * HH + y) * WW + x] = v0 > 0.f ? v0 : 0.f;
    out[((size_t)(b * CC + oc1) * HH + y) * WW + x] = v1 > 0.f ? v1 : 0.f;
  }
}

// ====================================================================
extern "C" void kernel_launch(void* const* d_in, const int* in_sizes, int n_in,
                              void* d_out, int out_size, void* d_ws, size_t ws_size,
                              hipStream_t stream) {
  const float* cur   = (const float*)d_in[0];
  const float* lut   = (const float*)d_in[1];
  const float* poses = (const float*)d_in[2];
  const float* Km    = (const float*)d_in[3];
  const float* invK  = (const float*)d_in[4];
  const float* cw    = (const float*)d_in[5];
  const float* cb    = (const float*)d_in[6];

  float* out      = (float*)d_out;
  float* outLow   = out + (size_t)BB * CC * HW;   // [B,H,W]
  float* outConf  = outLow + (size_t)BB * HW;     // [B,H,W]

  // workspace layout
  char* ws = (char*)d_ws;
  size_t off = 0;
  float* costRaw = (float*)(ws + off);            off += (size_t)BB * DD * HW * 4;
  unsigned short* catHi = (unsigned short*)(ws + off); off += (size_t)BB * HW * KIN * 2;
  unsigned short* catLo = (unsigned short*)(ws + off); off += (size_t)BB * HW * KIN * 2;
  unsigned short* wHi   = (unsigned short*)(ws + off); off += (size_t)KTOT * CC * 2;
  unsigned short* wLo   = (unsigned short*)(ws + off); off += (size_t)KTOT * CC * 2;

  int n1 = BB * DD * HH * WW;
  k1_plane_sweep<<<(n1 + 255) / 256, 256, 0, stream>>>(cur, lut, poses, Km, invK, costRaw);

  int n2 = BB * HW;
  k2_reduce_pack<<<(n2 + 255) / 256, 256, 0, stream>>>(costRaw, cur, outLow, outConf, catHi, catLo);

  int n3 = NKC * 8 * 32 * 16;
  k3_pack_weights<<<(n3 + 255) / 256, 256, 0, stream>>>(cw, wHi, wLo);

  int mTiles = (BB * HW) / 16;   // 960
  k4_conv_wmma<<<mTiles, 128, 0, stream>>>(catHi, catLo, wHi, wLo, cb, out);
}
